// SkipGram_31379031064645
// MI455X (gfx1250) — matrix operand order repaired and checked by reading, not compiled, not run on hardware
//
#include <hip/hip_runtime.h>

typedef __attribute__((ext_vector_type(16))) __bf16 v16bf;
typedef __attribute__((ext_vector_type(8)))  __bf16 v8bf;
typedef __attribute__((ext_vector_type(2)))  __bf16 v2bf;
typedef __attribute__((ext_vector_type(8)))  float  v8f;
typedef __attribute__((ext_vector_type(2)))  float  v2f;

constexpr int CTX   = 10;            // contexts per b
constexpr int NEG   = 20;            // negatives per context
constexpr int DIM   = 50;            // embedding dim (even -> pair staging is clean)
constexpr int ROWS  = CTX + CTX*NEG; // 210 scored words per b
constexpr int RPAD  = 224;           // rows padded to 14 tiles of 16
constexpr int TILES = RPAD / 16;     // 14
constexpr int STR   = 72;            // LDS row stride (bf16 elems) -> 144B, conflict-free
constexpr int SPR   = STR / 2;       // 36 bf16-pairs per row
constexpr int NPAIR = RPAD * SPR;    // 8064 staged pairs per block
constexpr int TPB   = 256;           // 8 waves (wave32)

__global__ __launch_bounds__(TPB) void sg_partials(
    const int*   __restrict__ target,
    const int*   __restrict__ contexts,
    const int*   __restrict__ neg_words,
    const float* __restrict__ W_in,
    const float* __restrict__ W_out,
    float*       __restrict__ partials)
{
  __shared__ __align__(16) __bf16 ldsW[RPAD * STR];
  __shared__ __align__(16) __bf16 tvs[64];
  __shared__ int   wIdx[RPAD];
  __shared__ float red[TPB];

  const int b   = blockIdx.x;
  const int tid = threadIdx.x;

  // Stage gathered word indices (-1 => zero padding row).
  if (tid < RPAD) {
    int w = -1;
    if (tid < CTX)       w = contexts [b * CTX + tid];
    else if (tid < ROWS) w = neg_words[b * (CTX * NEG) + (tid - CTX)];
    wIdx[tid] = w;
  }
  // Stage target vector (zero-padded to 64).
  if (tid < 64) {
    const int tg = target[b];
    float v = (tid < DIM) ? W_in[tg * DIM + tid] : 0.0f;
    tvs[tid] = (__bf16)v;
  }
  __syncthreads();

  // Gather 210 W_out rows into LDS as bf16, two elements at a time:
  // global_load_b64 -> packed f32->bf16 -> 4B LDS store. Every pair is
  // either fully valid (d2 <= 48) or fully padding, since DIM is even.
  for (int p = tid; p < NPAIR; p += TPB) {
    const int r  = p / SPR;
    const int d2 = (p - r * SPR) * 2;
    const int w  = wIdx[r];
    v2f v = {0.0f, 0.0f};
    if (w >= 0 && d2 < DIM) {
      const float2 g = *(const float2*)(W_out + w * DIM + d2);
      v.x = g.x; v.y = g.y;
    }
    v2bf pk;
    pk[0] = (__bf16)v.x;
    pk[1] = (__bf16)v.y;
    *(v2bf*)(&ldsW[r * STR + d2]) = pk;
  }
  __syncthreads();

  const int lane = tid & 31;
  const int wave = tid >> 5;
  const int n    = lane & 15;   // column / word-in-tile
  const int h    = lane >> 4;   // half-wave selector

  // A operand: tv replicated into all 16 rows (this code's contiguous K order:
  // lane m+16h holds K = h*16+[0..15]); rows identical => broadcast.
  v16bf a0, a1;
  #pragma unroll
  for (int j = 0; j < 8; ++j) {
    a0[j]     = tvs[h * 16 + j];
    a0[j + 8] = tvs[h * 16 + 8 + j];
    a1[j]     = tvs[32 + h * 16 + j];
    a1[j + 8] = tvs[32 + h * 16 + 8 + j];
  }

  float acc = 0.0f;
  for (int t = wave; t < TILES; t += TPB / 32) {
    // B operand (32x16 bf16): lane n+16h = column n, K = h*16 + [0..15].
    const __bf16* rp = &ldsW[(16 * t + n) * STR + h * 16];
    v8bf lo0 = *(const v8bf*)(rp);        // K-chunk 0, elems 0..7
    v8bf hi0 = *(const v8bf*)(rp + 8);    // K-chunk 0, elems 8..15
    v8bf lo1 = *(const v8bf*)(rp + 32);   // K-chunk 1 (d 32..)
    v8bf hi1 = *(const v8bf*)(rp + 40);
    v16bf b0 = __builtin_shufflevector(lo0, hi0, 0,1,2,3,4,5,6,7,8,9,10,11,12,13,14,15);
    v16bf b1 = __builtin_shufflevector(lo1, hi1, 0,1,2,3,4,5,6,7,8,9,10,11,12,13,14,15);

    v8f c = {};
    c = __builtin_amdgcn_wmma_f32_16x16x32_bf16(false, a0, false, b0, (short)0, c, false, false);
    c = __builtin_amdgcn_wmma_f32_16x16x32_bf16(false, a1, false, b1, (short)0, c, false, false);

    // D row M=0 lives in c[0], lanes 0..15 hold N=0..15 (all A-rows equal, so any M works).
    if (lane < 16) {
      const int w = 16 * t + n;
      if (w < ROWS) {
        const float s  = c[0];
        const float x  = (w < CTX) ? s : -s;           // negate negative-sample scores
        const float ls = fminf(x, 0.0f) - __logf(1.0f + __expf(-fabsf(x)));
        acc += ls;
      }
    }
  }

  // Deterministic block reduction.
  red[tid] = acc;
  __syncthreads();
  #pragma unroll
  for (int s = TPB / 2; s > 0; s >>= 1) {
    if (tid < s) red[tid] += red[tid + s];
    __syncthreads();
  }
  if (tid == 0) partials[b] = red[0];
}

__global__ __launch_bounds__(TPB) void sg_reduce(
    const float* __restrict__ partials, int nb, float scale, float* __restrict__ out)
{
  __shared__ float red[TPB];
  float s = 0.0f;
  for (int i = threadIdx.x; i < nb; i += TPB) s += partials[i];
  red[threadIdx.x] = s;
  __syncthreads();
  #pragma unroll
  for (int k = TPB / 2; k > 0; k >>= 1) {
    if (threadIdx.x < k) red[threadIdx.x] += red[threadIdx.x + k];
    __syncthreads();
  }
  if (threadIdx.x == 0) out[0] = -red[0] * scale;
}

extern "C" void kernel_launch(void* const* d_in, const int* in_sizes, int n_in,
                              void* d_out, int out_size, void* d_ws, size_t ws_size,
                              hipStream_t stream) {
  const int*   target    = (const int*)  d_in[0];
  const int*   contexts  = (const int*)  d_in[1];
  const int*   neg_words = (const int*)  d_in[2];
  const float* W_in      = (const float*)d_in[3];
  const float* W_out     = (const float*)d_in[4];

  const int B = in_sizes[0];              // 16384
  float* partials = (float*)d_ws;         // B floats of scratch

  sg_partials<<<B, TPB, 0, stream>>>(target, contexts, neg_words, W_in, W_out, partials);
  sg_reduce  <<<1, TPB, 0, stream>>>(partials, B, 1.0f / (float(B) * float(CTX)),
                                     (float*)d_out);
}
